// DecoderRNN_14912126452401
// MI455X (gfx1250) — matrix operand order repaired and checked by reading, not compiled, not run on hardware
//
#include <hip/hip_runtime.h>
#include <hip/hip_bf16.h>

typedef __attribute__((ext_vector_type(16))) _Float16 v16h;
typedef __attribute__((ext_vector_type(8)))  _Float16 v8h;
typedef __attribute__((ext_vector_type(8)))  float    v8f;

#define BB 128
#define LL 196
#define DD 512
#define HH 512
#define AAdim 512
#define EE 256
#define VV 32000
#define TT 24
#define KX 1280   /* E + D + H */
#define G4 2048   /* 4*H */

__device__ __forceinline__ float sigf(float x) { return 1.0f / (1.0f + __expf(-x)); }

// A fragment (16x32 f16): lane holds row; halves split K as {h*8..h*8+7, 16+h*8..}
__device__ __forceinline__ v16h ld_fragA(const _Float16* __restrict__ row, int k0, int half) {
    v8h lo = *(const v8h*)(row + k0 + half * 8);
    v8h hi = *(const v8h*)(row + k0 + 16 + half * 8);
    v16h f;
#pragma unroll
    for (int j = 0; j < 8; ++j) { f[j] = lo[j]; f[j + 8] = hi[j]; }
    return f;
}

// B fragment (32x16 f16): lane holds weight row (= B column); halves split K as {h*8..h*8+7, 16+h*8..}
__device__ __forceinline__ v16h ld_fragB(const _Float16* __restrict__ row, int k0, int half) {
    v8h lo = *(const v8h*)(row + k0 + half * 8);
    v8h hi = *(const v8h*)(row + k0 + 16 + half * 8);
    v16h f;
#pragma unroll
    for (int j = 0; j < 8; ++j) { f[j] = lo[j]; f[j + 8] = hi[j]; }
    return f;
}

#define WMMA_F16(AF, BF, CC) \
    __builtin_amdgcn_wmma_f32_16x16x32_f16(false, (AF), false, (BF), (short)0, (CC), false, false)

// ---------------------------------------------------------------------------
// f16-in / f32-accumulate WMMA GEMM:  C(M,N) = A(M,K) @ W(N,K)^T + bias
// Block = 256 threads = 8 waves arranged 2(M) x 4(N).
// Each wave computes 32(M) x 64(N): 2 A-frags x 4 B-frags -> 8 accumulators,
// with explicit double-buffered prefetch so loads overlap the WMMA burst.
// Block tile = 64 M x 256 N. M % 64 == 0, N % 256 == 0, K % 32 == 0 by design.
// ---------------------------------------------------------------------------
__global__ __launch_bounds__(256)
void wmma_gemm_kernel(const _Float16* __restrict__ A, int lda,
                      const _Float16* __restrict__ W, int ldb,
                      const float* __restrict__ bias,
                      float* __restrict__ C, int ldc,
                      int M, int N, int K) {
    const int lane = threadIdx.x & 31;
    const int wave = threadIdx.x >> 5;
    const int half = lane >> 4;     // 0: lanes 0-15, 1: lanes 16-31
    const int r16  = lane & 15;
    const int tm = blockIdx.y * 64  + (wave >> 2) * 32;   // 2 waves along M, 32 rows each
    const int tn = blockIdx.x * 256 + (wave & 3) * 64;    // 4 waves along N, 64 cols each
    if (tm >= M || tn >= N) return;

    const _Float16* arow0 = A + (size_t)(tm + r16) * lda;
    const _Float16* arow1 = A + (size_t)(tm + 16 + r16) * lda;
    const _Float16* wr0 = W + (size_t)(tn + r16) * ldb;
    const _Float16* wr1 = wr0 + (size_t)16 * ldb;
    const _Float16* wr2 = wr0 + (size_t)32 * ldb;
    const _Float16* wr3 = wr0 + (size_t)48 * ldb;

    v8f acc[2][4] = {};

    v16h af0 = ld_fragA(arow0, 0, half);
    v16h af1 = ld_fragA(arow1, 0, half);
    v16h bf0 = ld_fragB(wr0, 0, half);
    v16h bf1 = ld_fragB(wr1, 0, half);
    v16h bf2 = ld_fragB(wr2, 0, half);
    v16h bf3 = ld_fragB(wr3, 0, half);

    for (int k0 = 0; k0 < K; k0 += 32) {
        // prefetch next K-step into fresh registers (clamped: last iter is dead)
        const int kn = (k0 + 32 < K) ? (k0 + 32) : k0;
        v16h af0n = ld_fragA(arow0, kn, half);
        v16h af1n = ld_fragA(arow1, kn, half);
        v16h bf0n = ld_fragB(wr0, kn, half);
        v16h bf1n = ld_fragB(wr1, kn, half);
        v16h bf2n = ld_fragB(wr2, kn, half);
        v16h bf3n = ld_fragB(wr3, kn, half);

        acc[0][0] = WMMA_F16(af0, bf0, acc[0][0]);
        acc[0][1] = WMMA_F16(af0, bf1, acc[0][1]);
        acc[0][2] = WMMA_F16(af0, bf2, acc[0][2]);
        acc[0][3] = WMMA_F16(af0, bf3, acc[0][3]);
        acc[1][0] = WMMA_F16(af1, bf0, acc[1][0]);
        acc[1][1] = WMMA_F16(af1, bf1, acc[1][1]);
        acc[1][2] = WMMA_F16(af1, bf2, acc[1][2]);
        acc[1][3] = WMMA_F16(af1, bf3, acc[1][3]);

        af0 = af0n; af1 = af1n;
        bf0 = bf0n; bf1 = bf1n; bf2 = bf2n; bf3 = bf3n;
    }

#pragma unroll
    for (int mt = 0; mt < 2; ++mt) {
#pragma unroll
        for (int nt = 0; nt < 4; ++nt) {
            const int coln = tn + nt * 16 + r16;
            const float bv = bias ? bias[coln] : 0.0f;
#pragma unroll
            for (int r = 0; r < 8; ++r) {
                size_t row = (size_t)(tm + mt * 16 + r + half * 8);
                C[row * (size_t)ldc + coln] = acc[mt][nt][r] + bv;
            }
        }
    }
}

// ---------------------------------------------------------------------------
// Elementwise / prep kernels
// ---------------------------------------------------------------------------
__global__ void cvt_f16_kernel(const float* __restrict__ s, _Float16* __restrict__ d,
                               long long n) {
    long long i = (long long)blockIdx.x * blockDim.x + threadIdx.x;
    long long st = (long long)gridDim.x * blockDim.x;
    for (; i < n; i += st) d[i] = (_Float16)s[i];
}

__global__ void add_kernel(const float* __restrict__ a, const float* __restrict__ b,
                           float* __restrict__ o, int n) {
    int i = blockIdx.x * blockDim.x + threadIdx.x;
    if (i < n) o[i] = a[i] + b[i];
}

// W_cat (2048,1280) = [Wih (2048,768) | Whh (2048,512)] as f16
__global__ void pack_lstm_kernel(const float* __restrict__ wih, const float* __restrict__ whh,
                                 _Float16* __restrict__ wcat) {
    int i = blockIdx.x * blockDim.x + threadIdx.x;
    if (i >= G4 * KX) return;
    int r = i / KX, c = i % KX;
    float v = (c < EE + DD) ? wih[(size_t)r * (EE + DD) + c]
                            : whh[(size_t)r * HH + (c - EE - DD)];
    wcat[i] = (_Float16)v;
}

__global__ void mean_kernel(const float* __restrict__ a, float* __restrict__ m) {
    int i = blockIdx.x * blockDim.x + threadIdx.x;   // over B*D, d fastest
    if (i >= BB * DD) return;
    int d = i % DD, b = i / DD;
    const float* p = a + (size_t)b * LL * DD + d;
    float s = 0.0f;
    for (int l = 0; l < LL; ++l) s += p[(size_t)l * DD];
    m[i] = s * (1.0f / (float)LL);
}

__global__ void init_state_kernel(const float* __restrict__ mc,
                                  const float* __restrict__ hw, const float* __restrict__ hb,
                                  const float* __restrict__ cw, const float* __restrict__ cb,
                                  float* __restrict__ h, float* __restrict__ c) {
    int i = blockIdx.x * blockDim.x + threadIdx.x;   // over B*H, b fastest
    if (i >= BB * HH) return;
    int b = i & (BB - 1), j = i >> 7;
    const float* m  = mc + (size_t)b * DD;
    const float* wh = hw + (size_t)j * DD;
    const float* wc = cw + (size_t)j * DD;
    float sh = 0.0f, sc = 0.0f;
    for (int k = 0; k < DD; ++k) { float mv = m[k]; sh += mv * wh[k]; sc += mv * wc[k]; }
    h[(size_t)b * HH + j] = tanhf(sh + hb[j]);
    c[(size_t)b * HH + j] = tanhf(sc + cb[j]);
}

// hW = h @ (W_h + W_c)^T   (prev_h == h always in the reference scan)
__global__ void hW_kernel(const float* __restrict__ h, const float* __restrict__ whc,
                          float* __restrict__ hw) {
    int i = blockIdx.x * blockDim.x + threadIdx.x;   // over B*A, b fastest
    if (i >= BB * AAdim) return;
    int b = i & (BB - 1), j = i >> 7;
    const float* hr = h + (size_t)b * HH;
    const float* wr = whc + (size_t)j * HH;
    float s = 0.0f;
    for (int k = 0; k < HH; ++k) s += hr[k] * wr[k];
    hw[(size_t)b * AAdim + j] = s;
}

__global__ void score_kernel(const float* __restrict__ wa, const float* __restrict__ hw,
                             const float* __restrict__ vw, float* __restrict__ sc) {
    int i = blockIdx.x * blockDim.x + threadIdx.x;   // over B*L
    if (i >= BB * LL) return;
    int b = i / LL;
    const float* wr = wa + (size_t)i * AAdim;
    const float* hr = hw + (size_t)b * AAdim;
    float s = 0.0f;
    for (int k = 0; k < AAdim; ++k) s += vw[k] * tanhf(wr[k] + hr[k]);
    sc[i] = s;
}

// one block per batch element: softmax(196) -> alphas out, beta, gated context,
// and assemble xh = [emb_t | beta*context | h] in f16 for the gates GEMM
__global__ __launch_bounds__(256)
void attn_ctx_kernel(const float* __restrict__ sc, const float* __restrict__ a,
                     const float* __restrict__ h,  const float* __restrict__ bw,
                     const float* __restrict__ bb, const float* __restrict__ emb,
                     const int* __restrict__ cap,  int t,
                     _Float16* __restrict__ xh, float* __restrict__ alphas_out) {
    __shared__ float red[256];
    __shared__ float alpha[LL];
    __shared__ float s_beta;
    const int b = blockIdx.x, tid = threadIdx.x;

    float v = (tid < LL) ? sc[b * LL + tid] : -1e30f;
    red[tid] = v; __syncthreads();
    for (int s = 128; s > 0; s >>= 1) { if (tid < s) red[tid] = fmaxf(red[tid], red[tid + s]); __syncthreads(); }
    float mx = red[0]; __syncthreads();
    float e = (tid < LL) ? __expf(v - mx) : 0.0f;
    red[tid] = e; __syncthreads();
    for (int s = 128; s > 0; s >>= 1) { if (tid < s) red[tid] += red[tid + s]; __syncthreads(); }
    float inv = 1.0f / red[0];
    if (tid < LL) {
        float al = e * inv;
        alpha[tid] = al;
        alphas_out[(size_t)b * (TT - 1) * LL + (size_t)t * LL + tid] = al;
    }

    // beta = sigmoid(h . beta_w + beta_b)
    float pb = 0.0f;
    for (int k = tid; k < HH; k += 256) pb += h[(size_t)b * HH + k] * bw[k];
    __syncthreads();
    red[tid] = pb; __syncthreads();
    for (int s = 128; s > 0; s >>= 1) { if (tid < s) red[tid] += red[tid + s]; __syncthreads(); }
    if (tid == 0) s_beta = sigf(red[0] + bb[0]);
    __syncthreads();
    float beta = s_beta;

    // xh[:, 0:E) = embedding of captions[b, t]
    int tok = cap[b * TT + t];
    if (tid < EE) xh[(size_t)b * KX + tid] = (_Float16)emb[(size_t)tok * EE + tid];
    // xh[:, E:E+D) = beta * (alpha . a)
    for (int d = tid; d < DD; d += 256) {
        float s = 0.0f;
        const float* ap = a + (size_t)b * LL * DD + d;
        for (int l = 0; l < LL; ++l) s += alpha[l] * ap[(size_t)l * DD];
        xh[(size_t)b * KX + EE + d] = (_Float16)(beta * s);
    }
    // xh[:, E+D:E+D+H) = h
    for (int j = tid; j < HH; j += 256)
        xh[(size_t)b * KX + EE + DD + j] = (_Float16)h[(size_t)b * HH + j];
}

__global__ void lstm_kernel(const float* __restrict__ g, float* __restrict__ h,
                            float* __restrict__ c, _Float16* __restrict__ hh) {
    int i = blockIdx.x * blockDim.x + threadIdx.x;   // over B*H, j fastest
    if (i >= BB * HH) return;
    int j = i % HH, b = i / HH;
    const float* gr = g + (size_t)b * G4;
    float ig = sigf(gr[j]);
    float fg = sigf(gr[HH + j]);
    float gg = tanhf(gr[2 * HH + j]);
    float og = sigf(gr[3 * HH + j]);
    float cn = fg * c[i] + ig * gg;
    float hn = og * tanhf(cn);
    c[i] = cn;
    h[i] = hn;
    hh[i] = (_Float16)hn;
}

// ---------------------------------------------------------------------------
extern "C" void kernel_launch(void* const* d_in, const int* in_sizes, int n_in,
                              void* d_out, int out_size, void* d_ws, size_t ws_size,
                              hipStream_t stream) {
    (void)in_sizes; (void)n_in; (void)out_size; (void)ws_size;

    const float* a        = (const float*)d_in[0];
    const float* embed_w  = (const float*)d_in[1];
    const float* W_a      = (const float*)d_in[2];
    const float* W_h      = (const float*)d_in[3];
    const float* W_c      = (const float*)d_in[4];
    const float* v_w      = (const float*)d_in[5];
    const float* beta_w   = (const float*)d_in[6];
    const float* beta_b   = (const float*)d_in[7];
    const float* lstm_Wih = (const float*)d_in[8];
    const float* lstm_Whh = (const float*)d_in[9];
    const float* lstm_bih = (const float*)d_in[10];
    const float* lstm_bhh = (const float*)d_in[11];
    const float* fc_w     = (const float*)d_in[12];
    const float* fc_b     = (const float*)d_in[13];
    const float* init_h_w = (const float*)d_in[14];
    const float* init_h_b = (const float*)d_in[15];
    const float* init_c_w = (const float*)d_in[16];
    const float* init_c_b = (const float*)d_in[17];
    const int*   captions = (const int*)d_in[18];

    float* out        = (float*)d_out;
    float* alphas_out = out + (size_t)(TT - 1) * BB * VV;

    // workspace carve-out (256B aligned slices)
    char* ws = (char*)d_ws;
    size_t off = 0;
    auto give = [&](size_t bytes) -> void* {
        void* p = ws + off;
        off += (bytes + 255) & ~(size_t)255;
        return p;
    };
    _Float16* a_h    = (_Float16*)give((size_t)BB * LL * DD * 2);
    _Float16* wa_h   = (_Float16*)give((size_t)AAdim * DD * 2);
    _Float16* fcw_h  = (_Float16*)give((size_t)VV * HH * 2);
    _Float16* wcat_h = (_Float16*)give((size_t)G4 * KX * 2);
    float* wa_f  = (float*)give((size_t)BB * LL * AAdim * 4);
    float* whc   = (float*)give((size_t)AAdim * HH * 4);
    float* biasL = (float*)give((size_t)G4 * 4);
    float* meanc = (float*)give((size_t)BB * DD * 4);
    float* hbuf  = (float*)give((size_t)BB * HH * 4);
    float* cbuf  = (float*)give((size_t)BB * HH * 4);
    float* hWb   = (float*)give((size_t)BB * AAdim * 4);
    float* scb   = (float*)give((size_t)BB * LL * 4);
    _Float16* xh_h = (_Float16*)give((size_t)BB * KX * 2);
    _Float16* h_h  = (_Float16*)give((size_t)BB * HH * 2);
    float* gates = (float*)give((size_t)BB * G4 * 4);

    // ---- one-time prep ----
    cvt_f16_kernel<<<4096, 256, 0, stream>>>(a, a_h, (long long)BB * LL * DD);
    cvt_f16_kernel<<<1024, 256, 0, stream>>>(W_a, wa_h, (long long)AAdim * DD);
    cvt_f16_kernel<<<4096, 256, 0, stream>>>(fc_w, fcw_h, (long long)VV * HH);
    pack_lstm_kernel<<<(G4 * KX + 255) / 256, 256, 0, stream>>>(lstm_Wih, lstm_Whh, wcat_h);
    add_kernel<<<(G4 + 255) / 256, 256, 0, stream>>>(lstm_bih, lstm_bhh, biasL, G4);
    add_kernel<<<(AAdim * HH + 255) / 256, 256, 0, stream>>>(W_h, W_c, whc, AAdim * HH);
    mean_kernel<<<(BB * DD + 255) / 256, 256, 0, stream>>>(a, meanc);
    init_state_kernel<<<(BB * HH + 255) / 256, 256, 0, stream>>>(
        meanc, init_h_w, init_h_b, init_c_w, init_c_b, hbuf, cbuf);

    // wa = a @ W_a^T : (25088, 512) x (512, 512)
    {
        dim3 g(AAdim / 256, (BB * LL) / 64);
        wmma_gemm_kernel<<<g, 256, 0, stream>>>(a_h, DD, wa_h, DD, nullptr,
                                                wa_f, AAdim, BB * LL, AAdim, DD);
    }

    // ---- sequential decode ----
    for (int t = 0; t < TT - 1; ++t) {
        hW_kernel<<<(BB * AAdim + 255) / 256, 256, 0, stream>>>(hbuf, whc, hWb);
        score_kernel<<<(BB * LL + 255) / 256, 256, 0, stream>>>(wa_f, hWb, v_w, scb);
        attn_ctx_kernel<<<BB, 256, 0, stream>>>(scb, a, hbuf, beta_w, beta_b,
                                                embed_w, captions, t, xh_h, alphas_out);
        {   // gates = xh @ [Wih|Whh]^T + (bih+bhh) : (128,1280) x (1280,2048)
            dim3 g(G4 / 256, BB / 64);
            wmma_gemm_kernel<<<g, 256, 0, stream>>>(xh_h, KX, wcat_h, KX, biasL,
                                                    gates, G4, BB, G4, KX);
        }
        lstm_kernel<<<(BB * HH + 255) / 256, 256, 0, stream>>>(gates, hbuf, cbuf, h_h);
        {   // logits = h_new @ fc_w^T + fc_b : (128,512) x (512,32000)
            dim3 g(VV / 256, BB / 64);
            wmma_gemm_kernel<<<g, 256, 0, stream>>>(h_h, HH, fcw_h, HH, fc_b,
                                                    out + (size_t)t * BB * VV, VV,
                                                    BB, VV, HH);
        }
    }
}
